// V3MoE_59133109731522
// MI455X (gfx1250) — matrix operand order repaired and checked by reading, not compiled, not run on hardware
//
#include <hip/hip_runtime.h>

typedef __attribute__((ext_vector_type(16))) __bf16 bf16x16;
typedef __attribute__((ext_vector_type(8)))  float  f32x8;
typedef __attribute__((ext_vector_type(4)))  unsigned int u32x4;
typedef __attribute__((ext_vector_type(8)))  int i32x8;
typedef __attribute__((ext_vector_type(4)))  int i32x4;

#if defined(__gfx1250__) && __has_builtin(__builtin_amdgcn_tensor_load_to_lds)
#define HAS_TDM 1
#else
#define HAS_TDM 0
#endif

namespace {
constexpr int kT    = 4096;   // tokens = B*S
constexpr int kDim  = 768;
constexpr int kHid  = 512;
constexpr int kE    = 16;
constexpr int kG    = 8;
constexpr float kScale = 2.5f;

constexpr int MT   = 64;      // token tile (M)
constexpr int NT   = 64;      // output-col tile (N)
constexpr int KT   = 32;      // K step per bf16 WMMA
constexpr int LSTR = 40;      // LDS row stride in halfs: 64B data + 16B pad

// scratch layout (bytes)
constexpr size_t OFF_XBF   = 0;
constexpr size_t SZ_XBF    = (size_t)kT * kDim * 2;            // 6.0 MiB
constexpr size_t OFF_SHACT = OFF_XBF + SZ_XBF;
constexpr size_t SZ_SHACT  = (size_t)kT * kHid * 2;            // 4.0 MiB
constexpr size_t OFF_ACT   = OFF_SHACT + SZ_SHACT;
constexpr size_t SZ_ACT    = (size_t)kT * 4 * kHid * 2;        // 16.0 MiB
constexpr size_t OFF_CNT   = OFF_ACT + SZ_ACT;
constexpr size_t SZ_CNT    = 256;
constexpr size_t OFF_TOKK  = OFF_CNT + SZ_CNT;
constexpr size_t SZ_TOKK   = (size_t)kE * kT * 4;
constexpr size_t OFF_WL    = OFF_TOKK + SZ_TOKK;
constexpr size_t SZ_WL     = (size_t)kE * kT * 4;
// pre-transposed bf16 weights, [N][K] so B-fragments read K-contiguous
constexpr size_t OFF_WTU   = OFF_WL + SZ_WL;
constexpr size_t SZ_WEXP   = (size_t)kE * kDim * kHid * 2;     // 12 MiB each
constexpr size_t OFF_WTG   = OFF_WTU + SZ_WEXP;
constexpr size_t OFF_WTD   = OFF_WTG + SZ_WEXP;
constexpr size_t OFF_SWTU  = OFF_WTD + SZ_WEXP;
constexpr size_t SZ_WSH    = (size_t)kDim * kHid * 2;
constexpr size_t OFF_SWTG  = OFF_SWTU + SZ_WSH;
constexpr size_t OFF_SWTD  = OFF_SWTG + SZ_WSH;
} // namespace

__device__ __forceinline__ unsigned short f2bfbits(float f) {
  unsigned u = __builtin_bit_cast(unsigned, f);
  return (unsigned short)((u + 0x7FFFu + ((u >> 16) & 1u)) >> 16); // RNE
}
__device__ __forceinline__ __bf16 f2bf(float f) {
  unsigned short s = f2bfbits(f);
  return __builtin_bit_cast(__bf16, s);
}

union BFrag { bf16x16 v; uint4 q[2]; };

#if HAS_TDM
// Tensor Data Mover: 2D tile (tile0 x tile1, 2-byte elems) -> LDS with
// 16B pad every 64B (matches LSTR=40-half row stride). One issue per wave.
// 6-arg builtin form: (g0 v4u, g1 v8i, g2 v4i, g3 v4i, v8i, cpol)
__device__ __forceinline__ void tdm_load_2d(unsigned lds_byte_addr,
                                            const void* gaddr,
                                            int dim0, int dim1, int stride0,
                                            int tile0, int tile1) {
  unsigned long long ga = (unsigned long long)(size_t)gaddr;
  u32x4 g0;
  g0[0] = 1u;                                              // count=1 (valid D#)
  g0[1] = lds_byte_addr;                                   // LDS dest (bytes)
  g0[2] = (unsigned)(ga & 0xFFFFFFFFu);                    // global_addr[31:0]
  g0[3] = (unsigned)((ga >> 32) & 0x1FFFFFFu) | (2u << 30); // addr[56:32]|type=2
  i32x8 g1;
  g1[0] = (1 << 16) | (1 << 20) | (3 << 22) | (3 << 25);   // 2B, pad 4dw/16dw
  g1[1] = (dim0 & 0xFFFF) << 16;                           // tensor_dim0 lo
  g1[2] = ((dim0 >> 16) & 0xFFFF) | ((dim1 & 0xFFFF) << 16);
  g1[3] = ((dim1 >> 16) & 0xFFFF) | ((tile0 & 0xFFFF) << 16);
  g1[4] = (tile1 & 0xFFFF);                                // tile_dim1, dim2=0
  g1[5] = stride0;                                         // dim0_stride lo
  g1[6] = 0;
  g1[7] = 0;
  i32x4 z4 = {0, 0, 0, 0};
  i32x8 z8 = {0, 0, 0, 0, 0, 0, 0, 0};
  __builtin_amdgcn_tensor_load_to_lds(g0, g1, z4, z4, z8, 0);
}
__device__ __forceinline__ unsigned ldsAddr(const void* p) {
  return (unsigned)(size_t)p;   // flat LDS aperture: low 32 bits = LDS addr
}
#endif

// ---------------------------------------------------------------------------
// 1) prep: fp32 x -> bf16, zero per-expert counters
// ---------------------------------------------------------------------------
__global__ void prep_kernel(const float* __restrict__ x, __bf16* __restrict__ xbf,
                            int* __restrict__ cnt) {
  int i = blockIdx.x * blockDim.x + threadIdx.x;
  if (i < kT * kDim) xbf[i] = f2bf(x[i]);
  if (i < kE) cnt[i] = 0;
}

// ---------------------------------------------------------------------------
// 1b) weight transpose+pack: in [batch][K][N] f32 -> out [batch][N][K] bf16
//     (coalesced reads; packed 16B bf16 stores)
// ---------------------------------------------------------------------------
__global__ __launch_bounds__(256)
void wt_kernel(const float* __restrict__ in, __bf16* __restrict__ out,
               int K, int N) {
  const size_t ib = (size_t)blockIdx.y * K * N;
  int tid = blockIdx.x * 256 + threadIdx.x;   // over K*N/8
  int n  = tid % N;
  int kb = tid / N;                            // 8-element K block
  if (kb * 8 >= K) return;
  union { uint4 q; unsigned short h[8]; } pk;
#pragma unroll
  for (int i = 0; i < 8; ++i)
    pk.h[i] = f2bfbits(in[ib + (size_t)(kb * 8 + i) * N + n]);
  *(uint4*)(out + ib + (size_t)n * K + kb * 8) = pk.q;
}

// ---------------------------------------------------------------------------
// 2) routing: one block (128 thr) per token
// ---------------------------------------------------------------------------
__global__ __launch_bounds__(128)
void route_kernel(const float* __restrict__ x, const float* __restrict__ gate_w,
                  const float* __restrict__ gate_b, const float* __restrict__ bias,
                  int* __restrict__ cnt, int* __restrict__ tokk,
                  float* __restrict__ wl) {
  __shared__ float xrow[kDim];
  __shared__ float logits[kE];
  const int t = blockIdx.x;
  const int tid = threadIdx.x;

  for (int i = tid; i < kDim; i += 128) xrow[i] = x[(size_t)t * kDim + i];
  __syncthreads();

  const int e = tid >> 3, j = tid & 7;
  float s = 0.f;
  for (int d = j; d < kDim; d += 8) s += xrow[d] * gate_w[d * kE + e];
  s += __shfl_xor(s, 4, 8);
  s += __shfl_xor(s, 2, 8);
  s += __shfl_xor(s, 1, 8);
  if (j == 0) logits[e] = s + gate_b[e];
  __syncthreads();

  if (tid == 0) {
    float sc[kE], sb[kE];
    float mx = -1e30f;
    for (int i = 0; i < kE; ++i) mx = fmaxf(mx, logits[i]);
    float sum = 0.f;
    for (int i = 0; i < kE; ++i) { sc[i] = __expf(logits[i] - mx); sum += sc[i]; }
    float inv = 1.f / sum;
    for (int i = 0; i < kE; ++i) { sc[i] *= inv; sb[i] = sc[i] + bias[i]; }

    float gs[kG];
    for (int g = 0; g < kG; ++g) gs[g] = sb[2 * g] + sb[2 * g + 1];
    bool gsel[kG] = {};
    for (int k = 0; k < 4; ++k) {
      int best = -1; float bv = -3e38f;
      for (int g = 0; g < kG; ++g)
        if (!gsel[g] && gs[g] > bv) { bv = gs[g]; best = g; }
      gsel[best] = true;
    }
    bool taken[kE] = {};
    for (int k = 0; k < 4; ++k) {
      int best = -1; float bv = -3e38f;
      for (int i = 0; i < kE; ++i)
        if (gsel[i >> 1] && !taken[i] && sb[i] > bv) { bv = sb[i]; best = i; }
      taken[best] = true;
      float w = sc[best] * kScale;
      int pos = atomicAdd(&cnt[best], 1);
      tokk[best * kT + pos] = t * 4 + k;
      wl[best * kT + pos] = w;
    }
  }
}

// ---------------------------------------------------------------------------
// 3) fused up/gate GEMMs, double-buffered single-barrier pipeline.
//    B tiles streamed by TDM (tensor_load_to_lds) when available.
// ---------------------------------------------------------------------------
__global__ __launch_bounds__(256)
void upgate_kernel(const __bf16* __restrict__ xbf,
                   const __bf16* __restrict__ WtU, const float* __restrict__ bu,
                   const __bf16* __restrict__ WtG, const float* __restrict__ bg,
                   const int* __restrict__ cnt, const int* __restrict__ tokk,
                   const float* __restrict__ wl, __bf16* __restrict__ act,
                   int shared_mode) {
  __shared__ __align__(16) __bf16 lA [2][MT * LSTR];
  __shared__ __align__(16) __bf16 lBu[2][NT * LSTR];
  __shared__ __align__(16) __bf16 lBg[2][NT * LSTR];
  __shared__ int   rowSlot[MT];
  __shared__ float rowW[MT];

  const int e = shared_mode ? 0 : blockIdx.z;
  const int nrows = shared_mode ? kT : cnt[e];
  const int m0 = blockIdx.y * MT;
  if (m0 >= nrows) return;                    // block-uniform exit, EXEC stays full
  const int n0 = blockIdx.x * NT;

  const int tid = threadIdx.x;
  const int lane = tid & 31, wave = tid >> 5;
  const int wm = wave & 3, wn = wave >> 2;

  if (tid < MT) {
    int slot = -1; float w = 0.f;
    if (m0 + tid < nrows) {
      if (shared_mode) { slot = m0 + tid; w = 1.f; }
      else { slot = tokk[e * kT + m0 + tid]; w = wl[e * kT + m0 + tid]; }
    }
    rowSlot[tid] = slot; rowW[tid] = w;
  }
  __syncthreads();

  const __bf16* Wub = WtU + (size_t)e * kHid * kDim;  // [N=kHid][K=kDim]
  const __bf16* Wgb = WtG + (size_t)e * kHid * kDim;

  const int ar = tid >> 2, ac = tid & 3;      // 64 rows x 4 b128 chunks
  auto stageA = [&](int k0) -> uint4 {
    int slot = rowSlot[ar];
    uint4 v = make_uint4(0u, 0u, 0u, 0u);
    if (slot >= 0) {
      int tok = shared_mode ? slot : (slot >> 2);
      const __bf16* p = xbf + (size_t)tok * kDim + k0 + ac * 8;
      v = *(const uint4*)p;
      __builtin_prefetch((const char*)p + 64, 0, 1);   // next K-step of this row
    }
    return v;
  };

  uint4 rA = stageA(0);
#if HAS_TDM
  if (wave == 0) {
    tdm_load_2d(ldsAddr(&lBu[0][0]), Wub + (size_t)n0 * kDim,
                kDim, kHid, kDim, KT, NT);
    tdm_load_2d(ldsAddr(&lBg[0][0]), Wgb + (size_t)n0 * kDim,
                kDim, kHid, kDim, KT, NT);
  }
#else
  uint4 rBu, rBg;
  {
    const size_t off = (size_t)(n0 + ar) * kDim + ac * 8;
    rBu = *(const uint4*)(Wub + off);
    rBg = *(const uint4*)(Wgb + off);
  }
#endif

  f32x8 accU[2] = {}; f32x8 accG[2] = {};
  int buf = 0;
  for (int k0 = 0; k0 < kDim; k0 += KT, buf ^= 1) {
    *(uint4*)&lA[buf][ar * LSTR + ac * 8] = rA;
#if HAS_TDM
    if (wave == 0) __builtin_amdgcn_s_wait_tensorcnt(0);
#else
    *(uint4*)&lBu[buf][ar * LSTR + ac * 8] = rBu;
    *(uint4*)&lBg[buf][ar * LSTR + ac * 8] = rBg;
#endif
    __syncthreads();                           // tiles [buf] ready for all waves
    const int k1 = k0 + KT;
    if (k1 < kDim) {                           // prefetch next tiles into [buf^1]
      rA = stageA(k1);
#if HAS_TDM
      if (wave == 0) {
        tdm_load_2d(ldsAddr(&lBu[buf ^ 1][0]), Wub + (size_t)n0 * kDim + k1,
                    kDim, kHid, kDim, KT, NT);
        tdm_load_2d(ldsAddr(&lBg[buf ^ 1][0]), Wgb + (size_t)n0 * kDim + k1,
                    kDim, kHid, kDim, KT, NT);
      }
#else
      const size_t off = (size_t)(n0 + ar) * kDim + k1 + ac * 8;
      rBu = *(const uint4*)(Wub + off);
      rBg = *(const uint4*)(Wgb + off);
#endif
    }
    BFrag a;   // A-frag: lanes0-15 K{0-7,16-23}, lanes16-31 K{8-15,24-31}
    { int m = wm * 16 + (lane & 15); int c0 = (lane >> 4) * 8;
      a.q[0] = *(const uint4*)&lA[buf][m * LSTR + c0];
      a.q[1] = *(const uint4*)&lA[buf][m * LSTR + c0 + 16]; }
#pragma unroll
    for (int s = 0; s < 2; ++s) {
      int n = wn * 32 + s * 16 + (lane & 15); int c0 = (lane >> 4) * 8;
      BFrag b;
      b.q[0] = *(const uint4*)&lBu[buf][n * LSTR + c0];
      b.q[1] = *(const uint4*)&lBu[buf][n * LSTR + c0 + 16];
      accU[s] = __builtin_amdgcn_wmma_f32_16x16x32_bf16(false, a.v, false, b.v,
                                                        (short)0, accU[s], false, false);
      b.q[0] = *(const uint4*)&lBg[buf][n * LSTR + c0];
      b.q[1] = *(const uint4*)&lBg[buf][n * LSTR + c0 + 16];
      accG[s] = __builtin_amdgcn_wmma_f32_16x16x32_bf16(false, a.v, false, b.v,
                                                        (short)0, accG[s], false, false);
    }
  }

  const float* bub = bu + (shared_mode ? 0 : e * kHid);
  const float* bgb = bg + (shared_mode ? 0 : e * kHid);
#pragma unroll
  for (int s = 0; s < 2; ++s) {
    int hcol = n0 + wn * 32 + s * 16 + (lane & 15);
    float bU = bub[hcol], bG = bgb[hcol];
#pragma unroll
    for (int j = 0; j < 8; ++j) {
      int r = wm * 16 + j + 8 * (lane >> 4);   // C layout: row = j + 8*(lane>=16)
      int slot = rowSlot[r];
      if (slot >= 0) {
        float u = accU[s][j] + bU;
        float g = accG[s][j] + bG;
        float av = (u / (1.f + __expf(-u))) * g * rowW[r];
        act[(size_t)slot * kHid + hcol] = f2bf(av);
      }
    }
  }
}

// ---------------------------------------------------------------------------
// 4) shared-expert down-proj: y = shact @ sWd^T + sbd (initializes all of y)
//    Both A and B tiles are contiguous 2D -> TDM-streamed when available.
// ---------------------------------------------------------------------------
__global__ __launch_bounds__(256)
void shdown_kernel(const __bf16* __restrict__ A, const __bf16* __restrict__ Wt,
                   const float* __restrict__ bias, float* __restrict__ y) {
  __shared__ __align__(16) __bf16 lA[2][MT * LSTR];
  __shared__ __align__(16) __bf16 lB[2][NT * LSTR];
  const int m0 = blockIdx.y * MT, n0 = blockIdx.x * NT;
  const int tid = threadIdx.x, lane = tid & 31, wave = tid >> 5;
  const int wm = wave & 3, wn = wave >> 2;
  const int ar = tid >> 2, ac = tid & 3;
  f32x8 acc[2] = {};

#if HAS_TDM
  if (wave == 0) {
    tdm_load_2d(ldsAddr(&lA[0][0]), A + (size_t)m0 * kHid,
                kHid, kT, kHid, KT, MT);
    tdm_load_2d(ldsAddr(&lB[0][0]), Wt + (size_t)n0 * kHid,
                kHid, kDim, kHid, KT, NT);
  }
#else
  uint4 rA = *(const uint4*)(A  + (size_t)(m0 + ar) * kHid + ac * 8);
  uint4 rB = *(const uint4*)(Wt + (size_t)(n0 + ar) * kHid + ac * 8);
#endif

  int buf = 0;
  for (int k0 = 0; k0 < kHid; k0 += KT, buf ^= 1) {
#if HAS_TDM
    if (wave == 0) __builtin_amdgcn_s_wait_tensorcnt(0);
#else
    *(uint4*)&lA[buf][ar * LSTR + ac * 8] = rA;
    *(uint4*)&lB[buf][ar * LSTR + ac * 8] = rB;
#endif
    __syncthreads();
    const int k1 = k0 + KT;
    if (k1 < kHid) {
#if HAS_TDM
      if (wave == 0) {
        tdm_load_2d(ldsAddr(&lA[buf ^ 1][0]), A + (size_t)m0 * kHid + k1,
                    kHid, kT, kHid, KT, MT);
        tdm_load_2d(ldsAddr(&lB[buf ^ 1][0]), Wt + (size_t)n0 * kHid + k1,
                    kHid, kDim, kHid, KT, NT);
      }
#else
      rA = *(const uint4*)(A  + (size_t)(m0 + ar) * kHid + k1 + ac * 8);
      rB = *(const uint4*)(Wt + (size_t)(n0 + ar) * kHid + k1 + ac * 8);
#endif
    }
    BFrag a;
    { int m = wm * 16 + (lane & 15); int c0 = (lane >> 4) * 8;
      a.q[0] = *(const uint4*)&lA[buf][m * LSTR + c0];
      a.q[1] = *(const uint4*)&lA[buf][m * LSTR + c0 + 16]; }
#pragma unroll
    for (int s = 0; s < 2; ++s) {
      int n = wn * 32 + s * 16 + (lane & 15); int c0 = (lane >> 4) * 8;
      BFrag b;
      b.q[0] = *(const uint4*)&lB[buf][n * LSTR + c0];
      b.q[1] = *(const uint4*)&lB[buf][n * LSTR + c0 + 16];
      acc[s] = __builtin_amdgcn_wmma_f32_16x16x32_bf16(false, a.v, false, b.v,
                                                       (short)0, acc[s], false, false);
    }
  }
#pragma unroll
  for (int s = 0; s < 2; ++s) {
    int n = n0 + wn * 32 + s * 16 + (lane & 15);
    float bv = bias[n];
#pragma unroll
    for (int j = 0; j < 8; ++j) {
      int m = m0 + wm * 16 + j + 8 * (lane >> 4);
      y[(size_t)m * kDim + n] = acc[s][j] + bv;
    }
  }
}

// ---------------------------------------------------------------------------
// 5) MoE down-proj: y[t] += act_slot @ Wd^T[e] + w*bd[e] (atomic accumulate)
// ---------------------------------------------------------------------------
__global__ __launch_bounds__(256)
void down_kernel(const __bf16* __restrict__ act, const __bf16* __restrict__ WtD,
                 const float* __restrict__ bd, const int* __restrict__ cnt,
                 const int* __restrict__ tokk, const float* __restrict__ wl,
                 float* __restrict__ y) {
  __shared__ __align__(16) __bf16 lA[2][MT * LSTR];
  __shared__ __align__(16) __bf16 lB[2][NT * LSTR];
  __shared__ int   rowSlot[MT];
  __shared__ float rowW[MT];

  const int e = blockIdx.z;
  const int nrows = cnt[e];
  const int m0 = blockIdx.y * MT;
  if (m0 >= nrows) return;
  const int n0 = blockIdx.x * NT;

  const int tid = threadIdx.x, lane = tid & 31, wave = tid >> 5;
  const int wm = wave & 3, wn = wave >> 2;
  const int ar = tid >> 2, ac = tid & 3;

  if (tid < MT) {
    int slot = -1; float w = 0.f;
    if (m0 + tid < nrows) { slot = tokk[e * kT + m0 + tid]; w = wl[e * kT + m0 + tid]; }
    rowSlot[tid] = slot; rowW[tid] = w;
  }
  __syncthreads();

  const __bf16* Wb = WtD + (size_t)e * kDim * kHid;   // [N=kDim][K=kHid]
  auto stageA = [&](int k0) -> uint4 {
    int slot = rowSlot[ar];
    uint4 v = make_uint4(0u, 0u, 0u, 0u);
    if (slot >= 0) v = *(const uint4*)(act + (size_t)slot * kHid + k0 + ac * 8);
    return v;
  };

  uint4 rA = stageA(0);
#if HAS_TDM
  if (wave == 0)
    tdm_load_2d(ldsAddr(&lB[0][0]), Wb + (size_t)n0 * kHid,
                kHid, kDim, kHid, KT, NT);
#else
  uint4 rB = *(const uint4*)(Wb + (size_t)(n0 + ar) * kHid + ac * 8);
#endif

  f32x8 acc[2] = {};
  int buf = 0;
  for (int k0 = 0; k0 < kHid; k0 += KT, buf ^= 1) {
    *(uint4*)&lA[buf][ar * LSTR + ac * 8] = rA;
#if HAS_TDM
    if (wave == 0) __builtin_amdgcn_s_wait_tensorcnt(0);
#else
    *(uint4*)&lB[buf][ar * LSTR + ac * 8] = rB;
#endif
    __syncthreads();
    const int k1 = k0 + KT;
    if (k1 < kHid) {
      rA = stageA(k1);
#if HAS_TDM
      if (wave == 0)
        tdm_load_2d(ldsAddr(&lB[buf ^ 1][0]), Wb + (size_t)n0 * kHid + k1,
                    kHid, kDim, kHid, KT, NT);
#else
      rB = *(const uint4*)(Wb + (size_t)(n0 + ar) * kHid + k1 + ac * 8);
#endif
    }
    BFrag a;
    { int m = wm * 16 + (lane & 15); int c0 = (lane >> 4) * 8;
      a.q[0] = *(const uint4*)&lA[buf][m * LSTR + c0];
      a.q[1] = *(const uint4*)&lA[buf][m * LSTR + c0 + 16]; }
#pragma unroll
    for (int s = 0; s < 2; ++s) {
      int n = wn * 32 + s * 16 + (lane & 15); int c0 = (lane >> 4) * 8;
      BFrag b;
      b.q[0] = *(const uint4*)&lB[buf][n * LSTR + c0];
      b.q[1] = *(const uint4*)&lB[buf][n * LSTR + c0 + 16];
      acc[s] = __builtin_amdgcn_wmma_f32_16x16x32_bf16(false, a.v, false, b.v,
                                                       (short)0, acc[s], false, false);
    }
  }

  const float* bdb = bd + (size_t)e * kDim;
#pragma unroll
  for (int s = 0; s < 2; ++s) {
    int n = n0 + wn * 32 + s * 16 + (lane & 15);
    float bv = bdb[n];
#pragma unroll
    for (int j = 0; j < 8; ++j) {
      int r = wm * 16 + j + 8 * (lane >> 4);
      int slot = rowSlot[r];
      if (slot >= 0) {
        int tok = slot >> 2;
        atomicAdd(&y[(size_t)tok * kDim + n], acc[s][j] + rowW[r] * bv);
      }
    }
  }
}

// ---------------------------------------------------------------------------
extern "C" void kernel_launch(void* const* d_in, const int* in_sizes, int n_in,
                              void* d_out, int out_size, void* d_ws, size_t ws_size,
                              hipStream_t stream) {
  (void)in_sizes; (void)n_in; (void)out_size; (void)ws_size;
  const float* x      = (const float*)d_in[0];
  const float* gate_w = (const float*)d_in[1];
  const float* gate_b = (const float*)d_in[2];
  const float* bias   = (const float*)d_in[3];
  const float* Wu  = (const float*)d_in[4];
  const float* bu  = (const float*)d_in[5];
  const float* Wg  = (const float*)d_in[6];
  const float* bg  = (const float*)d_in[7];
  const float* Wd  = (const float*)d_in[8];
  const float* bd  = (const float*)d_in[9];
  const float* sWu = (const float*)d_in[10];
  const float* sbu = (const float*)d_in[11];
  const float* sWg = (const float*)d_in[12];
  const float* sbg = (const float*)d_in[13];
  const float* sWd = (const float*)d_in[14];
  const float* sbd = (const float*)d_in[15];
  float* y = (float*)d_out;

  char* ws = (char*)d_ws;
  __bf16* xbf   = (__bf16*)(ws + OFF_XBF);
  __bf16* shact = (__bf16*)(ws + OFF_SHACT);
  __bf16* act   = (__bf16*)(ws + OFF_ACT);
  int*    cnt   = (int*)(ws + OFF_CNT);
  int*    tokk  = (int*)(ws + OFF_TOKK);
  float*  wlst  = (float*)(ws + OFF_WL);
  __bf16* wtU   = (__bf16*)(ws + OFF_WTU);
  __bf16* wtG   = (__bf16*)(ws + OFF_WTG);
  __bf16* wtD   = (__bf16*)(ws + OFF_WTD);
  __bf16* swtU  = (__bf16*)(ws + OFF_SWTU);
  __bf16* swtG  = (__bf16*)(ws + OFF_SWTG);
  __bf16* swtD  = (__bf16*)(ws + OFF_SWTD);

  // 1) x -> bf16, zero counters
  prep_kernel<<<(kT * kDim + 255) / 256, 256, 0, stream>>>(x, xbf, cnt);
  // 1b) weights -> transposed bf16 [N][K]
  const int wblk = (kDim * kHid / 8 + 255) / 256;   // 192
  wt_kernel<<<dim3(wblk, kE), 256, 0, stream>>>(Wu,  wtU,  kDim, kHid);
  wt_kernel<<<dim3(wblk, kE), 256, 0, stream>>>(Wg,  wtG,  kDim, kHid);
  wt_kernel<<<dim3(wblk, kE), 256, 0, stream>>>(Wd,  wtD,  kHid, kDim);
  wt_kernel<<<dim3(wblk, 1),  256, 0, stream>>>(sWu, swtU, kDim, kHid);
  wt_kernel<<<dim3(wblk, 1),  256, 0, stream>>>(sWg, swtG, kDim, kHid);
  wt_kernel<<<dim3(wblk, 1),  256, 0, stream>>>(sWd, swtD, kHid, kDim);
  // 2) routing -> per-expert gathered lists
  route_kernel<<<kT, 128, 0, stream>>>(x, gate_w, gate_b, bias, cnt, tokk, wlst);
  // 3a) shared-expert up/gate
  upgate_kernel<<<dim3(kHid / NT, kT / MT, 1), 256, 0, stream>>>(
      xbf, swtU, sbu, swtG, sbg, cnt, tokk, wlst, shact, 1);
  // 3b) MoE up/gate (sparse, gathered)
  upgate_kernel<<<dim3(kHid / NT, kT / MT, kE), 256, 0, stream>>>(
      xbf, wtU, bu, wtG, bg, cnt, tokk, wlst, act, 0);
  // 4) shared-expert down-proj (writes/initializes y)
  shdown_kernel<<<dim3(kDim / NT, kT / MT), 256, 0, stream>>>(shact, swtD, sbd, y);
  // 5) MoE down-proj (atomic accumulate into y)
  down_kernel<<<dim3(kDim / NT, kT / MT, kE), 256, 0, stream>>>(
      act, wtD, bd, cnt, tokk, wlst, y);
}
